// OuterProductMeanUpdate_68384469286925
// MI455X (gfx1250) — matrix operand order repaired and checked by reading, not compiled, not run on hardware
//
#include <hip/hip_runtime.h>

#define D_SINGLE 512
#define RANK     32
#define D_PAIR   128
#define NROW     512
#define EPS      1e-5f

typedef __attribute__((ext_vector_type(16))) __bf16 v16bf;
typedef __attribute__((ext_vector_type(8)))  __bf16 v8bf;
typedef __attribute__((ext_vector_type(8)))  float  v8f;

// Split an fp32 value into hi/lo bf16 parts: x ~= hi + lo (error ~2^-16 rel).
__device__ __forceinline__ void split_bf16(float x, __bf16* hi, __bf16* lo) {
  __bf16 h = (__bf16)x;
  *hi = h;
  *lo = (__bf16)(x - (float)h);
}

// ---- CDNA5 WMMA per-lane operand loaders (contiguous k = 16h + s per lane half, wave32) ----
// A matrix 16x32 bf16: row M = lane%16.
//   lanes 0-15 : K = 0..15
//   lanes 16-31: K = 16..31
__device__ __forceinline__ v16bf load_A_lane(const __bf16* row, int lane) {
  const int o = (lane >> 4) << 4;  // 0 or 16
  v8bf lo = *reinterpret_cast<const v8bf*>(row + o);
  v8bf hi = *reinterpret_cast<const v8bf*>(row + o + 8);
  v16bf r;
#pragma unroll
  for (int t = 0; t < 8; ++t) { r[t] = lo[t]; r[t + 8] = hi[t]; }
  return r;
}

// B matrix 32x16 bf16: col N = lane%16; lanes 0-15 hold K=0..15, lanes 16-31 K=16..31.
// `col` points at 32 contiguous K values for this N (storage is [N][K] row-major).
__device__ __forceinline__ v16bf load_B_lane(const __bf16* col, int lane) {
  const int o = (lane >> 4) << 4;  // 0 or 16
  return *reinterpret_cast<const v16bf*>(col + o);
}

#define WMMA_BF16(A, B, C) \
  __builtin_amdgcn_wmma_f32_16x16x32_bf16(false, (A), false, (B), (short)0, (C), false, false)

// ---------------------------------------------------------------------------
// Kernel 1: shared-input LayerNorm + rank-32 projection (hi/lo bf16 outputs).
// grid (512, 2): y==0 -> a = LN(single_a)@w_left^T+b_left ; y==1 -> b side.
// ---------------------------------------------------------------------------
__global__ void __launch_bounds__(256)
proj_ln_kernel(const float* __restrict__ xa, const float* __restrict__ xb,
               const float* __restrict__ g,  const float* __restrict__ be,
               const float* __restrict__ wl, const float* __restrict__ bl,
               const float* __restrict__ wr, const float* __restrict__ br,
               __bf16* __restrict__ a_hi, __bf16* __restrict__ a_lo,
               __bf16* __restrict__ b_hi, __bf16* __restrict__ b_lo)
{
  __shared__ float redS[256];
  __shared__ float redQ[256];
  __shared__ float xn[D_SINGLE];
  __shared__ float part[256];

  const int  tid   = threadIdx.x;
  const int  row   = blockIdx.x;
  const bool right = (blockIdx.y != 0);

  const float* x    = (right ? xb : xa) + (size_t)row * D_SINGLE;
  const float* w    = right ? wr : wl;
  const float* bias = right ? br : bl;
  __bf16* dhi = (right ? b_hi : a_hi) + row * RANK;
  __bf16* dlo = (right ? b_lo : a_lo) + row * RANK;

  const float x0 = x[tid];
  const float x1 = x[tid + 256];
  redS[tid] = x0 + x1;
  redQ[tid] = x0 * x0 + x1 * x1;
  __syncthreads();
  for (int s = 128; s > 0; s >>= 1) {
    if (tid < s) { redS[tid] += redS[tid + s]; redQ[tid] += redQ[tid + s]; }
    __syncthreads();
  }
  const float mu  = redS[0] * (1.0f / D_SINGLE);
  const float var = redQ[0] * (1.0f / D_SINGLE) - mu * mu;
  const float rs  = rsqrtf(var + EPS);

  xn[tid]       = (x0 - mu) * rs * g[tid]       + be[tid];
  xn[tid + 256] = (x1 - mu) * rs * g[tid + 256] + be[tid + 256];
  __syncthreads();

  // 256-thread dot: thread = (chunk 0..7, r 0..31); 64 MACs each.
  const int r  = tid & 31;
  const int ch = tid >> 5;
  {
    float acc = 0.f;
    const float* wrow = w + r * D_SINGLE + ch * 64;
    const float* xs   = xn + ch * 64;
    for (int d = 0; d < 64; ++d) acc += xs[d] * wrow[d];
    part[ch * 32 + r] = acc;
  }
  __syncthreads();
  if (tid < RANK) {
    float acc = bias[tid];
#pragma unroll
    for (int cc = 0; cc < 8; ++cc) acc += part[cc * 32 + tid];
    split_bf16(acc, &dhi[tid], &dlo[tid]);
  }
}

// ---------------------------------------------------------------------------
// Kernel 2: fp32 -> split hi/lo bf16 for w_out viewed as [4096][32] (q=p*32+r).
// ---------------------------------------------------------------------------
__global__ void split_bf16_kernel(const float* __restrict__ in,
                                  __bf16* __restrict__ out_hi,
                                  __bf16* __restrict__ out_lo, int n)
{
  int i = blockIdx.x * blockDim.x + threadIdx.x;
  if (i < n) split_bf16(in[i], &out_hi[i], &out_lo[i]);
}

// ---------------------------------------------------------------------------
// Kernel 3: wbT[j][q] = sum_s b[j][s] * w_out_flat[q][s]   (512 x 4096 x 32)
// Split-bf16: 3 wmma per tile (bhi*whi + bhi*wlo + blo*whi), K=32 per wmma.
// grid (32, 32), 8 waves/block; output stored hi/lo bf16 [j][q] = [j][p][r],
// which is directly the B-operand storage for kernel 4.
// ---------------------------------------------------------------------------
__global__ void __launch_bounds__(256)
wb_gemm_kernel(const __bf16* __restrict__ b_hi, const __bf16* __restrict__ b_lo,
               const __bf16* __restrict__ w_hi, const __bf16* __restrict__ w_lo,
               __bf16* __restrict__ wbT_hi, __bf16* __restrict__ wbT_lo)
{
  const int wave = threadIdx.x >> 5;
  const int lane = threadIdx.x & 31;
  const int mt = blockIdx.y;                // j tile, 0..31
  const int nt = blockIdx.x * 8 + wave;     // q tile, 0..255
  const int n  = lane & 15;

  v16bf Ah = load_A_lane(b_hi + (mt * 16 + n) * RANK, lane);
  v16bf Al = load_A_lane(b_lo + (mt * 16 + n) * RANK, lane);
  v16bf Bh = load_B_lane(w_hi + (nt * 16 + n) * RANK, lane);
  v16bf Bl = load_B_lane(w_lo + (nt * 16 + n) * RANK, lane);

  v8f c = {};
  c = WMMA_BF16(Ah, Bh, c);
  c = WMMA_BF16(Ah, Bl, c);
  c = WMMA_BF16(Al, Bh, c);

  const int mbase = (lane >> 4) << 3;       // D: M = v + 8*(lane>=16), N = lane%16
#pragma unroll
  for (int v = 0; v < 8; ++v) {
    const size_t o = (size_t)(mt * 16 + mbase + v) * 4096 + nt * 16 + n;
    split_bf16(c[v], &wbT_hi[o], &wbT_lo[o]);
  }
}

// ---------------------------------------------------------------------------
// Kernel 4 (dominant, store-bound: 134 MB out):
// per j, OUT[i,p] = A[512x32] @ wbT[j]^T[32x128] + b_out, fused LN over p=128.
// Block: 64 i-rows x 128 p for one j. 8 waves: wave w -> m-tile w&3,
// n-tiles (w>>2)*4 .. +3  => 4 tiles * 3 split-bf16 wmma = 12 wmma per wave.
// ---------------------------------------------------------------------------
__global__ void __launch_bounds__(256)
pair_ln_kernel(const __bf16* __restrict__ a_hi, const __bf16* __restrict__ a_lo,
               const __bf16* __restrict__ wbT_hi, const __bf16* __restrict__ wbT_lo,
               const float* __restrict__ b_out, const float* __restrict__ lno_g,
               const float* __restrict__ lno_b, float* __restrict__ out)
{
  constexpr int TI  = 64;
  constexpr int PAD = 132;                  // 128+4 -> 528B rows, 16B aligned
  __shared__ float tile[TI * PAD];
  __shared__ float ps[256];
  __shared__ float pq[256];
  __shared__ float mu_s[TI];
  __shared__ float rs_s[TI];

  const int tid  = threadIdx.x;
  const int wave = tid >> 5;
  const int lane = tid & 31;
  const int i0   = blockIdx.x * TI;         // 8 i-tiles
  const int j    = blockIdx.y;              // 512
  const int mt   = wave & 3;
  const int ntb  = (wave >> 2) << 2;
  const int n    = lane & 15;
  const int mbase = (lane >> 4) << 3;

  v16bf Ah = load_A_lane(a_hi + (i0 + mt * 16 + n) * RANK, lane);
  v16bf Al = load_A_lane(a_lo + (i0 + mt * 16 + n) * RANK, lane);
  const __bf16* wbh = wbT_hi + (size_t)j * 4096;
  const __bf16* wbl = wbT_lo + (size_t)j * 4096;

#pragma unroll
  for (int t = 0; t < 4; ++t) {
    const int nt = ntb + t;
    v16bf Bh = load_B_lane(wbh + (nt * 16 + n) * RANK, lane);
    v16bf Bl = load_B_lane(wbl + (nt * 16 + n) * RANK, lane);
    v8f c = {};
    c = WMMA_BF16(Ah, Bh, c);
    c = WMMA_BF16(Ah, Bl, c);
    c = WMMA_BF16(Al, Bh, c);
    const int p = nt * 16 + n;
    const float bo = b_out[p];
#pragma unroll
    for (int v = 0; v < 8; ++v) {
      tile[(mt * 16 + mbase + v) * PAD + p] = c[v] + bo;
    }
  }
  __syncthreads();

  // LN stats: 4 threads per row, vectorized LDS reads (8 x float4 each).
  {
    const int r   = tid >> 2;
    const int seg = tid & 3;
    const float4* rp = reinterpret_cast<const float4*>(&tile[r * PAD + seg * 32]);
    float s = 0.f, q = 0.f;
#pragma unroll
    for (int t = 0; t < 8; ++t) {
      float4 v = rp[t];
      s += v.x + v.y + v.z + v.w;
      q += v.x * v.x + v.y * v.y + v.z * v.z + v.w * v.w;
    }
    ps[tid] = s;
    pq[tid] = q;
  }
  __syncthreads();
  if (tid < TI) {
    const float s = ps[tid * 4] + ps[tid * 4 + 1] + ps[tid * 4 + 2] + ps[tid * 4 + 3];
    const float q = pq[tid * 4] + pq[tid * 4 + 1] + pq[tid * 4 + 2] + pq[tid * 4 + 3];
    const float m   = s * (1.0f / D_PAIR);
    const float var = q * (1.0f / D_PAIR) - m * m;
    mu_s[tid] = m;
    rs_s[tid] = rsqrtf(var + EPS);
  }
  __syncthreads();

  // Output: float4 (global_store_b128), 512B contiguous per wave per store.
  float4* out4 = reinterpret_cast<float4*>(out);
  const float4* g4 = reinterpret_cast<const float4*>(lno_g);
  const float4* b4 = reinterpret_cast<const float4*>(lno_b);
  for (int idx = tid; idx < TI * (D_PAIR / 4); idx += 256) {
    const int il = idx >> 5;
    const int p4 = idx & 31;
    const float4 x = *reinterpret_cast<const float4*>(&tile[il * PAD + p4 * 4]);
    const float m  = mu_s[il];
    const float rg = rs_s[il];
    const float4 gg = g4[p4];
    const float4 bb = b4[p4];
    float4 o;
    o.x = (x.x - m) * rg * gg.x + bb.x;
    o.y = (x.y - m) * rg * gg.y + bb.y;
    o.z = (x.z - m) * rg * gg.z + bb.z;
    o.w = (x.w - m) * rg * gg.w + bb.w;
    out4[((size_t)(i0 + il) * NROW + j) * (D_PAIR / 4) + p4] = o;
  }
}

// ---------------------------------------------------------------------------
extern "C" void kernel_launch(void* const* d_in, const int* in_sizes, int n_in,
                              void* d_out, int out_size, void* d_ws, size_t ws_size,
                              hipStream_t stream)
{
  (void)in_sizes; (void)n_in; (void)out_size; (void)ws_size;
  const float* single_a = (const float*)d_in[0];
  const float* single_b = (const float*)d_in[1];
  const float* ln_g     = (const float*)d_in[2];
  const float* ln_b     = (const float*)d_in[3];
  const float* w_left   = (const float*)d_in[4];
  const float* b_left   = (const float*)d_in[5];
  const float* w_right  = (const float*)d_in[6];
  const float* b_right  = (const float*)d_in[7];
  const float* w_out    = (const float*)d_in[8];
  const float* b_out    = (const float*)d_in[9];
  const float* lno_g    = (const float*)d_in[10];
  const float* lno_b    = (const float*)d_in[11];
  float* out = (float*)d_out;

  char* ws = (char*)d_ws;
  size_t off = 0;
  __bf16* a_hi = (__bf16*)(ws + off); off += 32 << 10;   // 512*32 bf16
  __bf16* a_lo = (__bf16*)(ws + off); off += 32 << 10;
  __bf16* b_hi = (__bf16*)(ws + off); off += 32 << 10;
  __bf16* b_lo = (__bf16*)(ws + off); off += 32 << 10;
  __bf16* w_hi = (__bf16*)(ws + off); off += 256 << 10;  // 4096*32 bf16
  __bf16* w_lo = (__bf16*)(ws + off); off += 256 << 10;
  __bf16* wbT_hi = (__bf16*)(ws + off); off += (size_t)4 << 20; // 512*4096 bf16
  __bf16* wbT_lo = (__bf16*)(ws + off); off += (size_t)4 << 20;

  proj_ln_kernel<<<dim3(NROW, 2), 256, 0, stream>>>(
      single_a, single_b, ln_g, ln_b, w_left, b_left, w_right, b_right,
      a_hi, a_lo, b_hi, b_lo);

  split_bf16_kernel<<<dim3(512), 256, 0, stream>>>(w_out, w_hi, w_lo, 4096 * RANK);

  wb_gemm_kernel<<<dim3(32, 32), 256, 0, stream>>>(b_hi, b_lo, w_hi, w_lo,
                                                   wbT_hi, wbT_lo);

  pair_ln_kernel<<<dim3(8, NROW), 256, 0, stream>>>(
      a_hi, a_lo, wbT_hi, wbT_lo, b_out, lno_g, lno_b, out);
}
